// HyperConstru_50483045597672
// MI455X (gfx1250) — 8-bit matrix operand order repaired and checked by reading, compiled, not run on hardware
//
#include <hip/hip_runtime.h>
#include <hip/hip_bf16.h>

#define NNODES 8192
#define NEDGES 1024
#define ZD     256
#define KK     819

typedef float v2f __attribute__((ext_vector_type(2)));
typedef float v8f __attribute__((ext_vector_type(8)));
typedef int   v8i __attribute__((ext_vector_type(8)));

static __device__ __forceinline__ unsigned f2key(float f) {
    unsigned u = __float_as_uint(f);
    return u ^ ((unsigned)((int)u >> 31) | 0x80000000u);
}
static __device__ __forceinline__ float key2f(unsigned k) {
    unsigned u = (k & 0x80000000u) ? (k ^ 0x80000000u) : ~k;
    return __uint_as_float(u);
}

__global__ void simi_gemm(const float* __restrict__ x,
                          const int* __restrict__ y,
                          const float* __restrict__ hp,
                          float* __restrict__ simiT) {
    int wave = threadIdx.x >> 5;
    int lane = threadIdx.x & 31;
    int gwid = blockIdx.x * 8 + wave;
    int te = gwid & 63;
    int tn = gwid >> 6;
    int e0 = te << 4, n0 = tn << 4;

    int m  = lane & 15;
    int hi = lane >> 4;
    int cls = y[te];
    const float* arow = hp + (size_t)(cls * 16 + m) * ZD;
    const float* brow = x  + (size_t)(n0 + m) * ZD;
    int koff = hi * 2;

    v8f c = {};
    #pragma unroll 4
    for (int k0 = 0; k0 < ZD; k0 += 4) {
        v2f a, b;
        a.x = arow[k0 + koff]; a.y = arow[k0 + koff + 1];
        b.x = brow[k0 + koff]; b.y = brow[k0 + koff + 1];
        c = __builtin_amdgcn_wmma_f32_16x16x4_f32(false, a, false, b,
                                                  (short)0, c, false, false);
    }
    int mbase = hi * 8;
    float* out = simiT + (size_t)(e0 + mbase) * NNODES + n0 + m;
    #pragma unroll
    for (int v = 0; v < 8; ++v)
        out[(size_t)v * NNODES] = c[v];
}

__global__ void topk_kernel(const float* __restrict__ simiT,
                            float* __restrict__ thr,
                            float* __restrict__ invDE) {
    __shared__ unsigned hist[256];
    __shared__ unsigned sPrefix;
    __shared__ unsigned sRank;
    int e = blockIdx.x;
    int tid = threadIdx.x;
    const float* row = simiT + (size_t)e * NNODES;
    if (tid == 0) { sPrefix = 0u; sRank = KK; }
    __syncthreads();

    for (int pass = 3; pass >= 0; --pass) {
        hist[tid] = 0u;
        __syncthreads();
        int shift = pass * 8;
        unsigned himask = (pass == 3) ? 0u : (0xFFFFFFFFu << (shift + 8));
        unsigned pfx = sPrefix;
        for (int n = tid; n < NNODES; n += 256) {
            unsigned key = f2key(row[n]);
            if ((key & himask) == (pfx & himask))
                atomicAdd(&hist[(key >> shift) & 0xFFu], 1u);
        }
        __syncthreads();
        if (tid == 0) {
            unsigned cum = 0;
            for (int b = 255; b >= 0; --b) {
                unsigned cnt = hist[b];
                if (cum + cnt >= sRank) {
                    sRank -= cum;
                    sPrefix |= ((unsigned)b) << shift;
                    break;
                }
                cum += cnt;
            }
        }
        __syncthreads();
    }

    unsigned kkey = sPrefix;
    unsigned cnt = 0;
    for (int n = tid; n < NNODES; n += 256)
        cnt += (f2key(row[n]) > kkey) ? 1u : 0u;
    hist[tid] = cnt;
    __syncthreads();
    for (int s = 128; s > 0; s >>= 1) {
        if (tid < s) hist[tid] += hist[tid + s];
        __syncthreads();
    }
    if (tid == 0) {
        thr[e] = key2f(kkey);
        invDE[e] = 1.0f / (float)hist[0];
    }
}

__global__ void qscale_kernel(const float* __restrict__ invDE,
                              unsigned char* __restrict__ q,
                              float* __restrict__ wscale) {
    __shared__ float red[256];
    int tid = threadIdx.x;
    float m = 0.0f;
    for (int e = tid; e < NEDGES; e += 256) m = fmaxf(m, invDE[e]);
    red[tid] = m;
    __syncthreads();
    for (int s = 128; s > 0; s >>= 1) {
        if (tid < s) red[tid] = fmaxf(red[tid], red[tid + s]);
        __syncthreads();
    }
    float mx = red[0];
    if (tid == 0) *wscale = mx / 255.0f;
    for (int e = tid; e < NEDGES; e += 256) {
        float r = invDE[e] / mx * 255.0f;
        int qi = (int)(r + 0.5f);
        qi = qi < 0 ? 0 : (qi > 255 ? 255 : qi);
        q[e] = (unsigned char)qi;
    }
}

__global__ void build_h(const float* __restrict__ simiT,
                        const float* __restrict__ thr,
                        const unsigned char* __restrict__ q,
                        unsigned char* __restrict__ A8,
                        unsigned char* __restrict__ Hn) {
    __shared__ float sThr[64];
    __shared__ unsigned char sQ[64];
    __shared__ unsigned char tA[64][68];
    __shared__ unsigned char tH[64][68];
    int e0 = blockIdx.y * 64;
    int i0 = blockIdx.x * 64;
    int tid = threadIdx.x;
    if (tid < 64) { sThr[tid] = thr[e0 + tid]; sQ[tid] = q[e0 + tid]; }
    __syncthreads();
    for (int idx = tid; idx < 4096; idx += 256) {
        int er = idx >> 6, ic = idx & 63;
        float v = simiT[(size_t)(e0 + er) * NNODES + i0 + ic];
        bool bit = v > sThr[er];
        tA[er][ic] = bit ? sQ[er] : (unsigned char)0;
        tH[er][ic] = bit ? (unsigned char)1 : (unsigned char)0;
    }
    __syncthreads();
    for (int idx = tid; idx < 4096; idx += 256) {
        int ir = idx >> 6, ec = idx & 63;
        size_t o = (size_t)(i0 + ir) * NEDGES + e0 + ec;
        A8[o] = tA[ec][ir];
        Hn[o] = tH[ec][ir];
    }
}

__global__ void dv_kernel(const unsigned char* __restrict__ Hn,
                          float* __restrict__ dv2) {
    int i = blockIdx.x * 256 + threadIdx.x;
    const unsigned* row = (const unsigned*)(Hn + (size_t)i * NEDGES);
    unsigned s = 0;
    #pragma unroll 4
    for (int c = 0; c < NEDGES / 4; ++c) {
        unsigned w = row[c];
        s += (w & 0xFFu) + ((w >> 8) & 0xFFu) + ((w >> 16) & 0xFFu) + (w >> 24);
    }
    dv2[i] = 1.0f / sqrtf((float)s);
}

__global__ void g_gemm(const unsigned char* __restrict__ A8,
                       const unsigned char* __restrict__ Hn,
                       const float* __restrict__ dv2,
                       const float* __restrict__ wscale_p,
                       float* __restrict__ G) {
    int lane = threadIdx.x & 31;
    int wave = threadIdx.x >> 5;
    int wi = wave >> 2;
    int wj = wave & 3;
    int i_base = blockIdx.y * 64 + wi * 32;
    int j_base = blockIdx.x * 256 + wj * 64;

    int m  = lane & 15;
    int hi = lane >> 4;
    const unsigned char* aRow0 = A8 + (size_t)(i_base + m) * NEDGES;
    const unsigned char* aRow1 = aRow0 + (size_t)16 * NEDGES;
    int akoff = hi * 8;
    const unsigned char* bRow[4];
    #pragma unroll
    for (int t = 0; t < 4; ++t)
        bRow[t] = Hn + (size_t)(j_base + t * 16 + m) * NEDGES;
    int bkoff = hi * 8;

    v8i acc[2][4];
    #pragma unroll
    for (int it = 0; it < 2; ++it)
        #pragma unroll
        for (int jt = 0; jt < 4; ++jt)
            acc[it][jt] = (v8i){};

    for (int k0 = 0; k0 < NEDGES; k0 += 64) {
        __builtin_prefetch(aRow0 + k0 + 64, 0, 1);
        __builtin_prefetch(bRow[lane & 3] + k0 + 64, 0, 1);

        v8i a0, a1, b[4];
        #pragma unroll
        for (int j = 0; j < 8; ++j) {
            int ka = k0 + akoff + (j & 1) * 4 + ((j >> 1) & 1) * 16 + (j >> 2) * 32;
            a0[j] = *(const int*)(aRow0 + ka);
            a1[j] = *(const int*)(aRow1 + ka);
            int kb = k0 + bkoff + (j & 1) * 4 + ((j >> 1) & 1) * 16 + (j >> 2) * 32;
            #pragma unroll
            for (int t = 0; t < 4; ++t)
                b[t][j] = *(const int*)(bRow[t] + kb);
        }
        #pragma unroll
        for (int jt = 0; jt < 4; ++jt) {
            acc[0][jt] = __builtin_amdgcn_wmma_i32_16x16x64_iu8(
                false, a0, false, b[jt], acc[0][jt], false, false);
            acc[1][jt] = __builtin_amdgcn_wmma_i32_16x16x64_iu8(
                false, a1, false, b[jt], acc[1][jt], false, false);
        }
    }

    float ws = *wscale_p;
    int mbase = hi * 8;
    #pragma unroll
    for (int it = 0; it < 2; ++it) {
        #pragma unroll
        for (int jt = 0; jt < 4; ++jt) {
            int j = j_base + jt * 16 + m;
            float sj = dv2[j] * ws;
            #pragma unroll
            for (int v = 0; v < 8; ++v) {
                int i = i_base + it * 16 + mbase + v;
                G[(size_t)i * NNODES + j] = sj * dv2[i] * (float)acc[it][jt][v];
            }
        }
    }
}

extern "C" void kernel_launch(void* const* d_in, const int* in_sizes, int n_in,
                              void* d_out, int out_size, void* d_ws, size_t ws_size,
                              hipStream_t stream) {
    (void)in_sizes; (void)n_in; (void)out_size; (void)ws_size;
    const float* x  = (const float*)d_in[0];
    const int*   y  = (const int*)d_in[1];
    const float* hp = (const float*)d_in[2];

    float* G = (float*)d_out;
    float* hyper_x = G + (size_t)NNODES * NNODES;

    char* ws = (char*)d_ws;
    float*         simiT = (float*)ws;
    unsigned char* A8    = (unsigned char*)(ws + (size_t)32 * 1024 * 1024);
    unsigned char* Hn    = (unsigned char*)(ws + (size_t)40 * 1024 * 1024);
    float*         thr    = (float*)(ws + (size_t)48 * 1024 * 1024);
    float*         invDE  = thr + NEDGES;
    float*         dv2    = invDE + NEDGES;
    float*         wscale = dv2 + NNODES;
    unsigned char* q      = (unsigned char*)(wscale + 1);

    simi_gemm<<<4096, 256, 0, stream>>>(x, y, hp, simiT);
    topk_kernel<<<NEDGES, 256, 0, stream>>>(simiT, thr, invDE);
    qscale_kernel<<<1, 256, 0, stream>>>(invDE, q, wscale);
    dim3 gb(NNODES / 64, NEDGES / 64);
    build_h<<<gb, 256, 0, stream>>>(simiT, thr, q, A8, Hn);
    dv_kernel<<<NNODES / 256, 256, 0, stream>>>(Hn, dv2);
    dim3 gg(NNODES / 256, NNODES / 64);
    g_gemm<<<gg, 256, 0, stream>>>(A8, Hn, dv2, wscale, G);
    hipMemcpyAsync(hyper_x, x, (size_t)NNODES * ZD * sizeof(float),
                   hipMemcpyDeviceToDevice, stream);
}
